// GATEmbedding_8512625180969
// MI455X (gfx1250) — compile-verified, matrix operand order checked by reading, not run on hardware
//
#include <hip/hip_runtime.h>

// ---------------------------------------------------------------------------
// GATEmbedding forward for MI455X (gfx1250, wave32, WMMA).
// conv2/3/4 = implicit-GEMM with v_wmma_f32_16x16x32_f16 (f16 in, f32 acc).
// B operand staged in LDS in im2col fragment layout so the hot K-loop is
// 8 global dword (A) + 8 ds dword (B) loads per WMMA, all linear addresses.
//
// d_out layout (floats, concatenated return tuple):
//   [0,1024)      slices_hidden        (8,128)
//   [1024,3072)   image_feature_second (8,256)
//   [3072,3088)   prediction           (8,2)
//   [3088,3104)   slices_out           (8,2)
//   [3104,5992)   batch_attention      (8,19,19)
//
// Workspace layout (bytes):
//   0        : W2 f16 (64x288)      36864
//   36864    : W3 f16 (128x576)     147456
//   184320   : W4 f16 (128x1152)    294912
//   524288   : gavg f32 (128x128)   65536
//   589824   : feats f32 (128x128)  65536
//   1 MiB    : arena Q (33 MiB)   ping
//   34 MiB   : arena P (64 MiB)     pong
// total 98 MiB required.
// ---------------------------------------------------------------------------

typedef __attribute__((ext_vector_type(16))) _Float16 v16h;
typedef __attribute__((ext_vector_type(8)))  float    v8f;

// ---------------------------------------------------------------- helpers ---
__global__ void cvt_f32_f16(const float* __restrict__ src,
                            _Float16* __restrict__ dst, int n) {
  int i = blockIdx.x * 256 + threadIdx.x;
  if (i < n) dst[i] = (_Float16)src[i];
}

// --------------------------------------------------- conv1 + relu + pool ---
// in: (128 imgs, 1, 128, 128) f32 ; out pooled: (128, 32, 64, 64) f16
__global__ void conv1_relu_pool(const float* __restrict__ img,
                                const float* __restrict__ w1,
                                const float* __restrict__ b1,
                                _Float16* __restrict__ out) {
  int idx = blockIdx.x * 256 + threadIdx.x;
  if (idx >= 128 * 32 * 64 * 64) return;
  int px = idx & 63;
  int t  = idx >> 6;
  int py = t & 63; t >>= 6;
  int co = t & 31; t >>= 5;
  int im = t;
  const float* ip = img + (size_t)im * 128 * 128;
  float wv[9];
#pragma unroll
  for (int r = 0; r < 9; ++r) wv[r] = w1[co * 9 + r];
  float bb = b1[co];
  float mx = 0.f;  // relu outputs are >= 0
#pragma unroll
  for (int sy = 0; sy < 2; ++sy)
#pragma unroll
    for (int sx = 0; sx < 2; ++sx) {
      int yy = py * 2 + sy, xx = px * 2 + sx;
      float acc = bb;
#pragma unroll
      for (int dy = 0; dy < 3; ++dy) {
        int gy = yy + dy - 1;
        if (gy < 0 || gy >= 128) continue;
#pragma unroll
        for (int dx = 0; dx < 3; ++dx) {
          int gx = xx + dx - 1;
          if (gx < 0 || gx >= 128) continue;
          acc += wv[dy * 3 + dx] * ip[gy * 128 + gx];
        }
      }
      acc = acc > 0.f ? acc : 0.f;
      mx = fmaxf(mx, acc);
    }
  out[idx] = (_Float16)mx;
}

// ------------------------------------------- implicit-GEMM conv via WMMA ---
// C[cout][y][x] = relu(bias + sum_{cin,dy,dx} W*in), SAME 3x3.
// GEMM: M=COUT, N=HW*HW, K=CIN*9 (multiple of 32).
// One workgroup (4 waves) = one image row-tile of 16 x-positions; each wave
// owns cout tiles wave, wave+4, ... (16 rows each). B operand staged in LDS as im2col matrix Bmat[n][k]
// (Kpad = K+2 halves -> odd dword row stride, 16 lanes hit distinct banks),
// so every B fragment register is one linear ds_load_b32.
template <int CIN, int COUT, int HW>
__global__ __launch_bounds__(128) void conv_wmma_relu(
    const _Float16* __restrict__ in,   // [128][CIN][HW][HW]
    const _Float16* __restrict__ w16,  // [COUT][CIN*9]
    const float* __restrict__ bias,
    _Float16* __restrict__ out) {      // [128][COUT][HW][HW]
  constexpr int K    = CIN * 9;
  constexpr int Kpad = K + 2;
  constexpr int XT   = HW / 16;
  __shared__ _Float16 Bmat[16 * Kpad];  // [n=col 0..15][k 0..K)

  const int im  = blockIdx.y;
  const int tb  = blockIdx.x;
  const int y   = tb / XT;
  const int x0  = (tb - y * XT) * 16;
  const int tid = threadIdx.x;

  const _Float16* inp = in + (size_t)im * CIN * HW * HW;
  // im2col staging: decode (n,k)->(cin,dy,dx) once, off the hot loop.
  for (int idx = tid; idx < 16 * K; idx += 128) {
    int n   = idx / K;
    int k   = idx - n * K;
    int cin = k / 9;
    int rr  = k - cin * 9;
    int dy  = rr / 3;
    int dx  = rr - dy * 3;
    int gy  = y + dy - 1;
    int gx  = x0 + n + dx - 1;
    _Float16 v = (_Float16)0.f;
    if (gy >= 0 && gy < HW && gx >= 0 && gx < HW)
      v = inp[(cin * HW + gy) * HW + gx];
    Bmat[n * Kpad + k] = v;
  }
  __syncthreads();

  const int wave = tid >> 5;
  const int lane = tid & 31;
  const int hsel = lane >> 4;   // lane-half selector (wave32)
  const int row  = lane & 15;   // A: M row ; B: N col (local x)
  const _Float16* brow = Bmat + row * Kpad;

  for (int tm = wave; tm < COUT / 16; tm += 4) {
    const _Float16* wrow = w16 + (size_t)(tm * 16 + row) * K;
    v8f acc = {0.f, 0.f, 0.f, 0.f, 0.f, 0.f, 0.f, 0.f};
    for (int kk = 0; kk < K; kk += 32) {
      v16h afrag, bfrag;
      unsigned int* ap = reinterpret_cast<unsigned int*>(&afrag);
      unsigned int* bp = reinterpret_cast<unsigned int*>(&bfrag);
#pragma unroll
      for (int vi = 0; vi < 8; ++vi) {
        // 16-bit A/B K-pattern (ISA 7.12.2): K = (vi<4?0:16)+2*(vi&3)+8*hsel
        const int kb = kk + ((vi & 4) << 2) + 2 * (vi & 3) + 8 * hsel;
        ap[vi] = *reinterpret_cast<const unsigned int*>(wrow + kb);
        bp[vi] = *reinterpret_cast<const unsigned int*>(brow + kb);
      }
      acc = __builtin_amdgcn_wmma_f32_16x16x32_f16(
          false, afrag, false, bfrag, (short)0, acc, false, false);
    }
#pragma unroll
    for (int r8 = 0; r8 < 8; ++r8) {
      const int m = tm * 16 + r8 + 8 * hsel;  // D layout: VGPR r -> M=r(+8)
      float v = acc[r8] + bias[m];
      v = v > 0.f ? v : 0.f;
      out[((size_t)(im * COUT + m) * HW + y) * HW + x0 + row] = (_Float16)v;
    }
  }
}

// ----------------------------------------------------------- 2x2 maxpool ---
__global__ void maxpool2_f16(const _Float16* __restrict__ in,
                             _Float16* __restrict__ out, int C, int H,
                             int total) {
  int idx = blockIdx.x * 256 + threadIdx.x;
  if (idx >= total) return;
  int Ho = H >> 1;
  int px = idx % Ho;
  int t  = idx / Ho;
  int py = t % Ho; t /= Ho;
  int c  = t % C;  t /= C;
  int im = t;
  const _Float16* p = in + ((size_t)(im * C + c) * H + py * 2) * H + px * 2;
  float a = (float)p[0], b = (float)p[1], cc = (float)p[H], d = (float)p[H + 1];
  out[idx] = (_Float16)fmaxf(fmaxf(a, b), fmaxf(cc, d));
}

// ---------------------------------------------- spatial mean + cnn fc ------
__global__ void mean_hw(const _Float16* __restrict__ a4,
                        float* __restrict__ gavg) {
  int i = blockIdx.x * 256 + threadIdx.x;  // 128 imgs * 128 ch
  if (i >= 128 * 128) return;
  const _Float16* p = a4 + (size_t)i * 64;
  float s = 0.f;
  for (int t = 0; t < 64; ++t) s += (float)p[t];
  gavg[i] = s * (1.f / 64.f);
}

__global__ void fc_cnn_k(const float* __restrict__ gavg,
                         const float* __restrict__ w,
                         const float* __restrict__ bv,
                         float* __restrict__ feats) {
  int i = blockIdx.x * 256 + threadIdx.x;  // 128*128
  if (i >= 128 * 128) return;
  int im = i >> 7, e = i & 127;
  float s = bv[e];
  for (int c = 0; c < 128; ++c) s += gavg[im * 128 + c] * w[c * 128 + e];
  feats[i] = s;
}

__global__ void slices_hidden_k(const float* __restrict__ feats,
                                float* __restrict__ out0) {
  int b = blockIdx.x, e = threadIdx.x;  // 8 x 128
  float s = 0.f;
  for (int c = 0; c < 16; ++c) s += feats[((b * 16 + c) << 7) + e];
  out0[b * 128 + e] = s * (1.f / 16.f);
}

__global__ void slices_out_k(const float* __restrict__ sh,
                             const float* __restrict__ sfcw,
                             const float* __restrict__ sfcb,
                             float* __restrict__ out3) {
  int t = threadIdx.x;
  if (t >= 16) return;
  int b = t >> 1, o = t & 1;
  float s = sfcb[o];
  for (int e = 0; e < 128; ++e) s += sh[b * 128 + e] * sfcw[e * 2 + o];
  out3[b * 2 + o] = s;
}

// ---------------------------------------------------------------- GAT ------
// N=19, MID=3, mid=10, width=2. Only adj>0 matters (row-normalization in
// build_adj never affects outputs).
__device__ __forceinline__ bool adj_mask(int i, int j) {
  bool a;
  if (i == 0)       a = (j == 1) || (j > 8 && j < 12);
  else if (i == 1)  a = (j == 2) || (j > 6 && j < 14);
  else if (i == 2)  a = (j > 2);
  else if (i == 3)  a = (j == 4);
  else if (i == 18) a = (j == 17);
  else              a = (j == i - 1) || (j == i + 1);
  return a || (i == j);
}

__global__ __launch_bounds__(256) void gat_kernel(
    const float* __restrict__ feats, const float* __restrict__ attW,
    const float* __restrict__ atta, const float* __restrict__ outW,
    const float* __restrict__ outa, const float* __restrict__ llw,
    const float* __restrict__ llb, const float* __restrict__ fcw,
    const float* __restrict__ fcb, float* __restrict__ out) {
  const int b = blockIdx.x;
  const int tid = threadIdx.x;

  __shared__ _Float16 Wh[8 * 19 * 64];   // [k][n][h]
  __shared__ float    s1[152], s2[152];  // [k][n]
  __shared__ float    att[8 * 19 * 19];  // [k][i][j]
  __shared__ _Float16 xbuf[19 * 512];    // [i][k*64+h]
  __shared__ float    Wh2[19 * 128];     // [i][e]
  __shared__ float    o2s[19], o1s;
  __shared__ float    att2row[19];
  __shared__ float    if0[128];
  __shared__ float    if2v[256];

  // Wh = slices @ att_W   (nodes 0..2 are zero)
  for (int idx = tid; idx < 8 * 19 * 64; idx += 256) {
    int k = idx / (19 * 64);
    int rem = idx - k * 19 * 64;
    int n = rem >> 6;
    int h = rem & 63;
    float acc = 0.f;
    if (n >= 3) {
      const float* fp = feats + (size_t)(b * 16 + (n - 3)) * 128;
      const float* wp = attW + (size_t)k * 128 * 64 + h;
      for (int f = 0; f < 128; ++f) acc += fp[f] * wp[f * 64];
    }
    Wh[idx] = (_Float16)acc;
  }
  __syncthreads();

  // s1 = Wh.a1, s2 = Wh.a2
  for (int idx = tid; idx < 152; idx += 256) {
    int k = idx / 19;
    const _Float16* wh = Wh + idx * 64;
    const float* a1 = atta + k * 128;
    float sa = 0.f, sb = 0.f;
    for (int h = 0; h < 64; ++h) {
      float w = (float)wh[h];
      sa += w * a1[h];
      sb += w * a1[64 + h];
    }
    s1[idx] = sa;
    s2[idx] = sb;
  }
  __syncthreads();

  // masked leaky-relu logits
  for (int idx = tid; idx < 8 * 19 * 19; idx += 256) {
    int k = idx / 361;
    int rem = idx - k * 361;
    int i = rem / 19;
    int j = rem - i * 19;
    float e = s1[k * 19 + i] + s2[k * 19 + j];
    e = e >= 0.f ? e : 0.2f * e;
    att[idx] = adj_mask(i, j) ? e : -9.0e15f;
  }
  __syncthreads();

  // row softmax (in place)
  for (int idx = tid; idx < 152; idx += 256) {
    float* row = att + idx * 19;
    float m = row[0];
    for (int j = 1; j < 19; ++j) m = fmaxf(m, row[j]);
    float s = 0.f;
    for (int j = 0; j < 19; ++j) {
      float ev = expf(row[j] - m);
      row[j] = ev;
      s += ev;
    }
    float inv = 1.f / s;
    for (int j = 0; j < 19; ++j) row[j] *= inv;
  }
  __syncthreads();

  // batch_attention = mean over heads
  for (int idx = tid; idx < 361; idx += 256) {
    float s = 0.f;
    for (int k = 0; k < 8; ++k) s += att[k * 361 + idx];
    out[3104 + b * 361 + idx] = s * (1.f / 8.f);
  }

  // h = att @ Wh ; elu ; transpose -> xbuf[i][k*64+h]
  for (int idx = tid; idx < 19 * 512; idx += 256) {
    int i  = idx >> 9;
    int kh = idx & 511;
    int k  = kh >> 6;
    int hh = kh & 63;
    const float* arow = att + (k * 19 + i) * 19;
    const _Float16* wcol = Wh + k * 19 * 64 + hh;
    float s = 0.f;
    for (int j = 0; j < 19; ++j) s += arow[j] * (float)wcol[j * 64];
    s = s > 0.f ? s : (expf(s) - 1.f);
    xbuf[idx] = (_Float16)s;
  }
  __syncthreads();

  // Wh2 = x @ out_W  (512 -> 128)
  for (int idx = tid; idx < 19 * 128; idx += 256) {
    int i = idx >> 7;
    int e = idx & 127;
    const _Float16* xp = xbuf + i * 512;
    float s = 0.f;
    for (int q = 0; q < 512; ++q) s += (float)xp[q] * outW[q * 128 + e];
    Wh2[idx] = s;
  }
  __syncthreads();

  // second attention: only row 0 is consumed downstream
  if (tid < 19) {
    float s = 0.f;
    const float* w = Wh2 + tid * 128;
    for (int e = 0; e < 128; ++e) s += w[e] * outa[128 + e];
    o2s[tid] = s;
  }
  if (tid == 32) {
    float s = 0.f;
    for (int e = 0; e < 128; ++e) s += Wh2[e] * outa[e];
    o1s = s;
  }
  __syncthreads();
  if (tid == 0) {
    float ev[19], m = -1e30f;
    for (int j = 0; j < 19; ++j) {
      float e = o1s + o2s[j];
      e = e >= 0.f ? e : 0.2f * e;
      e = adj_mask(0, j) ? e : -9.0e15f;
      ev[j] = e;
      m = fmaxf(m, e);
    }
    float s = 0.f;
    for (int j = 0; j < 19; ++j) { ev[j] = expf(ev[j] - m); s += ev[j]; }
    float inv = 1.f / s;
    for (int j = 0; j < 19; ++j) att2row[j] = ev[j] * inv;
  }
  __syncthreads();

  // image_feature[0] = elu(att2row @ Wh2)
  if (tid < 128) {
    float s = 0.f;
    for (int j = 0; j < 19; ++j) s += att2row[j] * Wh2[j * 128 + tid];
    s = s > 0.f ? s : (expf(s) - 1.f);
    if0[tid] = s;
  }
  __syncthreads();

  // ll: relu(if0 @ ll_w + ll_b) -> output 1
  {
    float s = llb[tid];
    for (int e = 0; e < 128; ++e) s += if0[e] * llw[e * 256 + tid];
    s = fmaxf(s, 0.f);
    if2v[tid] = s;
    out[1024 + b * 256 + tid] = s;
  }
  __syncthreads();

  // prediction -> output 2
  if (tid < 2) {
    float s = fcb[tid];
    for (int o = 0; o < 256; ++o) s += if2v[o] * fcw[o * 2 + tid];
    out[3072 + b * 2 + tid] = s;
  }
}

// --------------------------------------------------------------- driver ----
extern "C" void kernel_launch(void* const* d_in, const int* in_sizes, int n_in,
                              void* d_out, int out_size, void* d_ws,
                              size_t ws_size, hipStream_t stream) {
  (void)in_sizes; (void)n_in; (void)out_size; (void)ws_size;
  const float* img      = (const float*)d_in[0];
  const float* w1       = (const float*)d_in[1];
  const float* b1       = (const float*)d_in[2];
  const float* w2       = (const float*)d_in[3];
  const float* b2       = (const float*)d_in[4];
  const float* w3       = (const float*)d_in[5];
  const float* b3       = (const float*)d_in[6];
  const float* w4       = (const float*)d_in[7];
  const float* b4       = (const float*)d_in[8];
  const float* fc_cnn_w = (const float*)d_in[9];
  const float* fc_cnn_b = (const float*)d_in[10];
  const float* att_W    = (const float*)d_in[11];
  const float* att_a    = (const float*)d_in[12];
  const float* out_W    = (const float*)d_in[13];
  const float* out_a    = (const float*)d_in[14];
  const float* ll_w     = (const float*)d_in[15];
  const float* ll_b     = (const float*)d_in[16];
  const float* fc_w     = (const float*)d_in[17];
  const float* fc_b     = (const float*)d_in[18];
  const float* sfc_w    = (const float*)d_in[19];
  const float* sfc_b    = (const float*)d_in[20];

  float* out = (float*)d_out;
  char*  ws  = (char*)d_ws;

  _Float16* W2h   = (_Float16*)(ws + 0);
  _Float16* W3h   = (_Float16*)(ws + 36864);
  _Float16* W4h   = (_Float16*)(ws + 184320);
  float*    gavg  = (float*)(ws + 524288);
  float*    feats = (float*)(ws + 589824);
  _Float16* Q     = (_Float16*)(ws + (size_t)(1u << 20));       // 33 MiB arena
  _Float16* P     = (_Float16*)(ws + (size_t)35651584u);        // 64 MiB arena

  // f32 -> f16 weight blobs (as [Cout][Cin*9] row-major, same memory order)
  cvt_f32_f16<<<(18432 + 255) / 256, 256, 0, stream>>>(w2, W2h, 18432);
  cvt_f32_f16<<<(73728 + 255) / 256, 256, 0, stream>>>(w3, W3h, 73728);
  cvt_f32_f16<<<(147456 + 255) / 256, 256, 0, stream>>>(w4, W4h, 147456);

  // conv1 + relu + pool : img -> Q (128,32,64,64) f16
  conv1_relu_pool<<<65536, 256, 0, stream>>>(img, w1, b1, Q);

  // conv2 (WMMA) : Q -> P (128,64,64,64) ; pool -> Q (128,64,32,32)
  conv_wmma_relu<32, 64, 64><<<dim3(256, 128), 128, 0, stream>>>(Q, W2h, b2, P);
  maxpool2_f16<<<(8388608 + 255) / 256, 256, 0, stream>>>(P, Q, 64, 64, 8388608);

  // conv3 (WMMA) : Q -> P (128,128,32,32) ; pool -> Q (128,128,16,16)
  conv_wmma_relu<64, 128, 32><<<dim3(64, 128), 128, 0, stream>>>(Q, W3h, b3, P);
  maxpool2_f16<<<(4194304 + 255) / 256, 256, 0, stream>>>(P, Q, 128, 32, 4194304);

  // conv4 (WMMA) : Q -> P (128,128,16,16) ; pool -> Q (128,128,8,8)
  conv_wmma_relu<128, 128, 16><<<dim3(16, 128), 128, 0, stream>>>(Q, W4h, b4, P);
  maxpool2_f16<<<(1048576 + 255) / 256, 256, 0, stream>>>(P, Q, 128, 16, 1048576);

  // spatial mean, cnn fc, slice heads
  mean_hw<<<64, 256, 0, stream>>>(Q, gavg);
  fc_cnn_k<<<64, 256, 0, stream>>>(gavg, fc_cnn_w, fc_cnn_b, feats);
  slices_hidden_k<<<8, 128, 0, stream>>>(feats, out);
  slices_out_k<<<1, 16, 0, stream>>>(out, sfc_w, sfc_b, out + 3088);

  // GAT (both attention levels + heads), one block per batch element
  gat_kernel<<<8, 256, 0, stream>>>(feats, att_W, att_a, out_W, out_a, ll_w,
                                    ll_b, fc_w, fc_b, out);
}
